// ContinousThoughtMachineDyn_25194278159065
// MI455X (gfx1250) — matrix operand order repaired and checked by reading, not compiled, not run on hardware
//
#include <hip/hip_runtime.h>

typedef __attribute__((ext_vector_type(16))) _Float16 v16h;
typedef __attribute__((ext_vector_type(8)))  _Float16 v8h;
typedef __attribute__((ext_vector_type(8)))  float    v8f;

union V16 { v16h v; v8h h[2]; };

constexpr int Bn   = 64;
constexpr int Tt   = 16;
constexpr int DF   = 512;
constexpr int Sn   = 196;
constexpr int DIN  = 512;
constexpr int DMn  = 1024;
constexpr int MEMN = 25;
constexpr int HNn  = 32;
constexpr int DOUTn= 512;

// ---------------------------------------------------------------------------
// Register-blocked WMMA GEMM: C[M,N] = A[M,K] (f16 row-major) @ B + bias.
// B supplied transposed: Bt[n*K + k] (f16).
// Wave tile: 32 (M) x 64 (N) = 2x4 WMMA accumulators -> 8 wmma per 12 b128
// loads (A reused 4x, B reused 2x). Block = 4 waves -> 32 x 256 macro-tile.
// A-operand lane layout (16-bit A 16x32): lane L holds row M=L&15;
//   elements 0..7 -> K = kb..kb+7, elements 8..15 -> K = kb+16..kb+23,
//   kb = k0 + 8*(L>>4).
// B-operand: lane L holds col N=L&15, K in the same order as A (same kb).
// C/D layout: VGPR v -> row m0 + v + 8*(L>>4), col n0 + (L&15).
// ---------------------------------------------------------------------------
__global__ void gemm_wmma_f16(const _Float16* __restrict__ A, long strideA, int lda,
                              const _Float16* __restrict__ Bt,
                              const float* __restrict__ bias,
                              float* __restrict__ C, long strideC,
                              _Float16* __restrict__ Ch, long strideCh,
                              int M, int N, int K)
{
    const int lane = threadIdx.x & 31;
    const int wave = threadIdx.x >> 5;
    const int lg   = lane >> 4;
    const int m0   = blockIdx.x * 32;
    const int n0   = blockIdx.y * 256 + wave * 64;
    const int bat  = blockIdx.z;

    const _Float16* Ab = A + (long)bat * strideA;

    const _Float16* arow[2];
    #pragma unroll
    for (int mi = 0; mi < 2; ++mi) {
        int r = m0 + mi * 16 + (lane & 15);
        if (r >= M) r = M - 1;                    // clamp; stores are guarded
        arow[mi] = Ab + (size_t)r * lda;
    }
    const _Float16* brow[4];
    #pragma unroll
    for (int j = 0; j < 4; ++j)
        brow[j] = Bt + (size_t)(n0 + j * 16 + (lane & 15)) * K;

    v8f acc[2][4] = {};
    for (int k0 = 0; k0 < K; k0 += 32) {
        const int ka = k0 + (lg << 3);
        const int kb = k0 + (lg << 3);
        V16 a[2], b[4];
        #pragma unroll
        for (int mi = 0; mi < 2; ++mi) {
            a[mi].h[0] = *(const v8h*)(arow[mi] + ka);
            a[mi].h[1] = *(const v8h*)(arow[mi] + ka + 16);
        }
        #pragma unroll
        for (int j = 0; j < 4; ++j) {
            b[j].h[0] = *(const v8h*)(brow[j] + kb);
            b[j].h[1] = *(const v8h*)(brow[j] + kb + 16);
        }
        __builtin_prefetch(arow[0] + ka + 128, 0, 1);   // global_prefetch_b8
        __builtin_prefetch(arow[1] + ka + 128, 0, 1);
        __builtin_prefetch(brow[(k0 >> 5) & 3] + kb + 128, 0, 1);
        #pragma unroll
        for (int mi = 0; mi < 2; ++mi)
            #pragma unroll
            for (int j = 0; j < 4; ++j)
                acc[mi][j] = __builtin_amdgcn_wmma_f32_16x16x32_f16(
                    false, a[mi].v, false, b[j].v, (short)0, acc[mi][j], false, false);
    }

    #pragma unroll
    for (int j = 0; j < 4; ++j) {
        const int col = n0 + j * 16 + (lane & 15);
        const float bv = bias ? bias[col] : 0.0f;
        #pragma unroll
        for (int mi = 0; mi < 2; ++mi) {
            #pragma unroll
            for (int v = 0; v < 8; ++v) {
                const int r = m0 + mi * 16 + v + (lg << 3);
                if (r < M) {
                    const float val = acc[mi][j][v] + bv;
                    if (C)  C [(long)bat * strideC  + (size_t)r * N + col] = val;
                    if (Ch) Ch[(long)bat * strideCh + (size_t)r * N + col] = (_Float16)val;
                }
            }
        }
    }
}

// Wt[n*K+k] = (f16) W[k*N+n]   (transpose + downconvert weights once per call)
__global__ void wt_convert(const float* __restrict__ W, _Float16* __restrict__ Wt,
                           int K, int N)
{
    const long idx = (long)blockIdx.x * blockDim.x + threadIdx.x;
    if (idx >= (long)K * N) return;
    const int n = (int)(idx / K), k = (int)(idx % K);
    Wt[idx] = (_Float16)W[(size_t)k * N + n];
}

// xth[b][s][f] = (f16) x[b][t][f][s]
__global__ void x_transpose(const float* __restrict__ x, _Float16* __restrict__ xth, int t)
{
    const long idx = (long)blockIdx.x * blockDim.x + threadIdx.x;
    if (idx >= (long)Bn * Sn * DF) return;
    const int f = (int)(idx % DF);
    const int s = (int)((idx / DF) % Sn);
    const int b = (int)(idx / ((long)DF * Sn));
    xth[idx] = (_Float16)x[(((long)b * Tt + t) * DF + f) * Sn + s];
}

// LayerNorm over rows of [R, C] (f16 in, f16 out)
__global__ void ln_rows_h(const _Float16* __restrict__ in, int C,
                          const float* __restrict__ g, const float* __restrict__ bb,
                          _Float16* __restrict__ outh)
{
    __shared__ float red[256];
    const long row = blockIdx.x;
    const _Float16* p = in + row * C;
    float s = 0.f, s2 = 0.f;
    for (int c = threadIdx.x; c < C; c += 256) {
        const float v = (float)p[c]; s += v; s2 += v * v;
    }
    red[threadIdx.x] = s; __syncthreads();
    for (int o = 128; o > 0; o >>= 1) { if (threadIdx.x < o) red[threadIdx.x] += red[threadIdx.x + o]; __syncthreads(); }
    const float mean = red[0] / C; __syncthreads();
    red[threadIdx.x] = s2; __syncthreads();
    for (int o = 128; o > 0; o >>= 1) { if (threadIdx.x < o) red[threadIdx.x] += red[threadIdx.x + o]; __syncthreads(); }
    const float var = red[0] / C - mean * mean;
    const float inv = rsqrtf(var + 1e-5f);
    for (int c = threadIdx.x; c < C; c += 256)
        outh[row * C + c] = (_Float16)(((float)p[c] - mean) * inv * g[c] + bb[c]);
}

// Fused GLU + LayerNorm over [B, 2048] -> [B, 1024] f32
__global__ void glu_ln(const float* __restrict__ g2, const float* __restrict__ gam,
                       const float* __restrict__ bet, float* __restrict__ outf)
{
    __shared__ float buf[DMn];
    __shared__ float red[256];
    const int b = blockIdx.x;
    float s = 0.f, s2 = 0.f;
    for (int c = threadIdx.x; c < DMn; c += 256) {
        const float a = g2[b * 2048 + c];
        const float z = g2[b * 2048 + 1024 + c];
        const float v = a * (1.f / (1.f + expf(-z)));
        buf[c] = v; s += v; s2 += v * v;
    }
    red[threadIdx.x] = s; __syncthreads();
    for (int o = 128; o > 0; o >>= 1) { if (threadIdx.x < o) red[threadIdx.x] += red[threadIdx.x + o]; __syncthreads(); }
    const float mean = red[0] / DMn; __syncthreads();
    red[threadIdx.x] = s2; __syncthreads();
    for (int o = 128; o > 0; o >>= 1) { if (threadIdx.x < o) red[threadIdx.x] += red[threadIdx.x + o]; __syncthreads(); }
    const float var = red[0] / DMn - mean * mean;
    const float inv = rsqrtf(var + 1e-5f);
    for (int c = threadIdx.x; c < DMn; c += 256)
        outf[b * DMn + c] = (buf[c] - mean) * inv * gam[c] + bet[c];
}

// act[b,n]=init_act[n]; aa=ba=0; ao=init pairwise; bo=1
__global__ void init_state(const float* __restrict__ init_act,
                           const int* __restrict__ li_o, const int* __restrict__ ri_o,
                           float* __restrict__ act,
                           float* __restrict__ aa, float* __restrict__ ba,
                           float* __restrict__ ao, float* __restrict__ bo)
{
    const int idx = blockIdx.x * blockDim.x + threadIdx.x;
    if (idx >= Bn * DMn) return;
    const int b = idx / DMn, n = idx % DMn;
    act[idx] = init_act[n];
    if (n < 512) {
        const int j = b * 512 + n;
        aa[j] = 0.f; ba[j] = 0.f;
        ao[j] = init_act[li_o[n]] * init_act[ri_o[n]];
        bo[j] = 1.f;
    }
}

// num = r*num + act[li]*act[ri]; den = r*den + 1; sync = num/sqrt(den)
__global__ void sync_update(const float* __restrict__ act,
                            const int* __restrict__ li, const int* __restrict__ ri,
                            const float* __restrict__ decay,
                            float* __restrict__ num, float* __restrict__ den,
                            _Float16* __restrict__ sync_h)
{
    const int idx = blockIdx.x * blockDim.x + threadIdx.x;
    if (idx >= Bn * 512) return;
    const int b = idx / 512, j = idx % 512;
    const float r = expf(-fminf(fmaxf(decay[j], 0.f), 15.f));
    const float nu = r * num[idx] + act[b * DMn + li[j]] * act[b * DMn + ri[j]];
    const float de = r * den[idx] + 1.f;
    num[idx] = nu; den[idx] = de;
    sync_h[idx] = (_Float16)(nu * rsqrtf(de));
}

// 1-query MHA core: block = (b*8 + h), 64 threads; f16 output feeds o-proj GEMM
__global__ void attention(const float* __restrict__ q,      // [B,512]
                          const _Float16* __restrict__ kh,  // [B,S,512]
                          const _Float16* __restrict__ vh,  // [B,S,512]
                          _Float16* __restrict__ outh)      // [B,512] f16
{
    __shared__ float sc[Sn];
    __shared__ float red[64];
    const int b = blockIdx.x >> 3, h = blockIdx.x & 7;
    const float* qp = q + b * 512 + h * 64;
    const _Float16* kb = kh + ((long)b * Sn) * 512 + h * 64;
    for (int s = threadIdx.x; s < Sn; s += 64) {
        const _Float16* kr = kb + (size_t)s * 512;
        float d = 0.f;
        #pragma unroll 8
        for (int e = 0; e < 64; ++e) d += qp[e] * (float)kr[e];
        sc[s] = d * 0.125f;                       // 1/sqrt(64)
    }
    __syncthreads();
    float mx = -1e30f;
    for (int s = threadIdx.x; s < Sn; s += 64) mx = fmaxf(mx, sc[s]);
    red[threadIdx.x] = mx; __syncthreads();
    for (int o = 32; o > 0; o >>= 1) { if (threadIdx.x < o) red[threadIdx.x] = fmaxf(red[threadIdx.x], red[threadIdx.x + o]); __syncthreads(); }
    mx = red[0]; __syncthreads();
    float sm = 0.f;
    for (int s = threadIdx.x; s < Sn; s += 64) { const float e = expf(sc[s] - mx); sc[s] = e; sm += e; }
    red[threadIdx.x] = sm; __syncthreads();
    for (int o = 32; o > 0; o >>= 1) { if (threadIdx.x < o) red[threadIdx.x] += red[threadIdx.x + o]; __syncthreads(); }
    sm = red[0]; __syncthreads();
    const int d = threadIdx.x;                    // 0..63
    const _Float16* vb = vh + ((long)b * Sn) * 512 + h * 64 + d;
    float acc = 0.f;
    for (int s = 0; s < Sn; ++s) acc += sc[s] * (float)vb[(size_t)s * 512];
    outh[b * 512 + h * 64 + d] = (_Float16)(acc / sm);
}

// pre_h = [f16(attn) | f16(act)]  [B, 1536]
__global__ void concat_pre(const float* __restrict__ attn, const float* __restrict__ act,
                           _Float16* __restrict__ pre)
{
    const int idx = blockIdx.x * blockDim.x + threadIdx.x;
    if (idx >= Bn * 1536) return;
    const int b = idx / 1536, c = idx % 1536;
    const float v = (c < 512) ? attn[b * 512 + c] : act[b * DMn + (c - 512)];
    pre[idx] = (_Float16)v;
}

// Per-neuron 2-layer MLP over memory window. Block = neuron n, thread = batch b.
// Window (T<MEM, no wrap): m<=23-t -> init_hist[n, m+t+1]; else preact tick (t-24+m).
__global__ void nlm_kernel(const float* __restrict__ init_hist,  // [DM,25]
                           const float* __restrict__ ph,         // [T][B*DM]
                           int t,
                           const float* __restrict__ W1, const float* __restrict__ b1,
                           const float* __restrict__ W2, const float* __restrict__ b2,
                           float* __restrict__ act)
{
    const int n = blockIdx.x;      // 0..1023
    const int b = threadIdx.x;     // 0..63
    float vals[MEMN];
    #pragma unroll
    for (int m = 0; m < MEMN; ++m) {
        if (m <= 23 - t) vals[m] = init_hist[n * MEMN + m + t + 1];
        else             vals[m] = ph[(long)(t - 24 + m) * (Bn * DMn) + b * DMn + n];
    }
    float accv = 0.f;
    const float* wbase = W1 + (size_t)n * MEMN * HNn;
    for (int j = 0; j < HNn; ++j) {
        float hj = b1[n * HNn + j];
        #pragma unroll
        for (int m = 0; m < MEMN; ++m) hj += vals[m] * wbase[m * HNn + j];
        hj = fmaxf(hj, 0.f);
        accv += hj * W2[n * HNn + j];
    }
    act[b * DMn + n] = accv + b2[n];
}

// Write predictions[B,DOUT,T] and certainties[B,2,T] (entropy of softmax)
__global__ void finalize(const float* __restrict__ pred, float* __restrict__ out, int t)
{
    __shared__ float red[512];
    const int b = blockIdx.x;
    const int d = threadIdx.x;   // 0..511
    const float v = pred[b * DOUTn + d];
    out[((long)b * DOUTn + d) * Tt + t] = v;
    red[d] = v; __syncthreads();
    for (int o = 256; o > 0; o >>= 1) { if (d < o) red[d] = fmaxf(red[d], red[d + o]); __syncthreads(); }
    const float mx = red[0]; __syncthreads();
    red[d] = expf(v - mx); __syncthreads();
    for (int o = 256; o > 0; o >>= 1) { if (d < o) red[d] += red[d + o]; __syncthreads(); }
    const float lse = logf(red[0]); __syncthreads();
    const float lp = v - mx - lse;
    red[d] = expf(lp) * lp; __syncthreads();
    for (int o = 256; o > 0; o >>= 1) { if (d < o) red[d] += red[d + o]; __syncthreads(); }
    if (d == 0) {
        const float ne = -red[0] / logf((float)DOUTn);
        const long base = (long)Bn * DOUTn * Tt;
        out[base + ((long)b * 2 + 0) * Tt + t] = ne;
        out[base + ((long)b * 2 + 1) * Tt + t] = 1.f - ne;
    }
}

// ---------------------------------------------------------------------------
extern "C" void kernel_launch(void* const* d_in, const int* in_sizes, int n_in,
                              void* d_out, int out_size, void* d_ws, size_t ws_size,
                              hipStream_t stream)
{
    (void)in_sizes; (void)n_in; (void)out_size; (void)ws_size;
    const float* x        = (const float*)d_in[0];
    const float* Wqp      = (const float*)d_in[1];
    const float* bqp      = (const float*)d_in[2];
    const float* Wkv      = (const float*)d_in[3];
    const float* bkv      = (const float*)d_in[4];
    const float* g_kv     = (const float*)d_in[5];
    const float* b_kv     = (const float*)d_in[6];
    const float* Wq       = (const float*)d_in[7];
    const float* bq       = (const float*)d_in[8];
    const float* Wk       = (const float*)d_in[9];
    const float* bk       = (const float*)d_in[10];
    const float* Wv       = (const float*)d_in[11];
    const float* bv       = (const float*)d_in[12];
    const float* Wo       = (const float*)d_in[13];
    const float* bo       = (const float*)d_in[14];
    const float* Wsyn     = (const float*)d_in[15];
    const float* bsyn     = (const float*)d_in[16];
    const float* g_syn    = (const float*)d_in[17];
    const float* b_syn    = (const float*)d_in[18];
    const float* W1       = (const float*)d_in[19];
    const float* b1       = (const float*)d_in[20];
    const float* W2       = (const float*)d_in[21];
    const float* b2       = (const float*)d_in[22];
    const float* init_act = (const float*)d_in[23];
    const float* init_hist= (const float*)d_in[24];
    const float* decay_a  = (const float*)d_in[25];
    const float* decay_o  = (const float*)d_in[26];
    const float* Wout     = (const float*)d_in[27];
    const float* bout     = (const float*)d_in[28];
    const int*   li_a     = (const int*)d_in[29];
    const int*   ri_a     = (const int*)d_in[30];
    const int*   li_o     = (const int*)d_in[31];
    const int*   ri_o     = (const int*)d_in[32];

    char* ws = (char*)d_ws;
    size_t off = 0;
    auto alloc = [&](size_t bytes) -> char* {
        char* p = ws + off;
        off += (bytes + 255) & ~(size_t)255;
        return p;
    };

    // f16 transposed weights (persistent per call)
    _Float16* Wkv_t  = (_Float16*)alloc((size_t)512 * 512 * 2);
    _Float16* Wq_t   = (_Float16*)alloc((size_t)512 * 512 * 2);
    _Float16* Wk_t   = (_Float16*)alloc((size_t)512 * 512 * 2);
    _Float16* Wv_t   = (_Float16*)alloc((size_t)512 * 512 * 2);
    _Float16* Wo_t   = (_Float16*)alloc((size_t)512 * 512 * 2);
    _Float16* Wqp_t  = (_Float16*)alloc((size_t)512 * 512 * 2);
    _Float16* Wout_t = (_Float16*)alloc((size_t)512 * 512 * 2);
    _Float16* Wsyn_t = (_Float16*)alloc((size_t)1536 * 2048 * 2);
    // per-tick f16 tensors
    _Float16* xth     = (_Float16*)alloc((size_t)Bn * Sn * DF * 2);
    _Float16* kvpre_h = (_Float16*)alloc((size_t)Bn * Sn * DIN * 2);
    _Float16* kv_h    = (_Float16*)alloc((size_t)Bn * Sn * DIN * 2);
    _Float16* kh      = (_Float16*)alloc((size_t)Bn * Sn * DIN * 2);
    _Float16* vh      = (_Float16*)alloc((size_t)Bn * Sn * DIN * 2);
    _Float16* syncA_h = (_Float16*)alloc((size_t)Bn * 512 * 2);
    _Float16* q1_h    = (_Float16*)alloc((size_t)Bn * 512 * 2);
    _Float16* attno_h = (_Float16*)alloc((size_t)Bn * 512 * 2);
    _Float16* pre_h   = (_Float16*)alloc((size_t)Bn * 1536 * 2);
    _Float16* syncO_h = (_Float16*)alloc((size_t)Bn * 512 * 2);
    // f32 tensors
    float* q2_f    = (float*)alloc((size_t)Bn * 512 * 4);
    float* attnp_f = (float*)alloc((size_t)Bn * 512 * 4);   // after o-proj
    float* g_f     = (float*)alloc((size_t)Bn * 2048 * 4);
    float* ph      = (float*)alloc((size_t)Tt * Bn * DMn * 4);  // pre-act history
    float* act     = (float*)alloc((size_t)Bn * DMn * 4);
    float* aaN     = (float*)alloc((size_t)Bn * 512 * 4);
    float* aaD     = (float*)alloc((size_t)Bn * 512 * 4);
    float* aoN     = (float*)alloc((size_t)Bn * 512 * 4);
    float* aoD     = (float*)alloc((size_t)Bn * 512 * 4);
    float* pred_f  = (float*)alloc((size_t)Bn * 512 * 4);

    auto WT = [&](const float* W, _Float16* Wt, int K, int N) {
        const long tot = (long)K * N;
        wt_convert<<<(tot + 255) / 256, 256, 0, stream>>>(W, Wt, K, N);
    };
    WT(Wkv,  Wkv_t,  512, 512);
    WT(Wq,   Wq_t,   512, 512);
    WT(Wk,   Wk_t,   512, 512);
    WT(Wv,   Wv_t,   512, 512);
    WT(Wo,   Wo_t,   512, 512);
    WT(Wqp,  Wqp_t,  512, 512);
    WT(Wout, Wout_t, 512, 512);
    WT(Wsyn, Wsyn_t, 1536, 2048);

    init_state<<<(Bn * DMn + 255) / 256, 256, 0, stream>>>(init_act, li_o, ri_o,
                                                           act, aaN, aaD, aoN, aoD);

    auto GEMM = [&](const _Float16* A, long sA, int lda, const _Float16* Bt,
                    const float* bias, float* C, long sC, _Float16* Ch, long sCh,
                    int M, int N, int K, int batch) {
        dim3 grid((M + 31) / 32, N / 256, batch);   // 4 waves x (32x64) per block
        gemm_wmma_f16<<<grid, 128, 0, stream>>>(A, sA, lda, Bt, bias, C, sC, Ch, sCh, M, N, K);
    };

    const long SB = (long)Sn * DIN;   // per-batch stride of [S,512] tensors

    for (int t = 0; t < Tt; ++t) {
        // kv = LN(x_t^T @ Wkv + bkv)
        x_transpose<<<((long)Bn * Sn * DF + 255) / 256, 256, 0, stream>>>(x, xth, t);
        GEMM(xth, SB, DF, Wkv_t, bkv, nullptr, 0, kvpre_h, SB, Sn, DIN, DF, Bn);
        ln_rows_h<<<Bn * Sn, 256, 0, stream>>>(kvpre_h, DIN, g_kv, b_kv, kv_h);
        // K,V projections
        GEMM(kv_h, SB, DIN, Wk_t, bk, nullptr, 0, kh, SB, Sn, DIN, DIN, Bn);
        GEMM(kv_h, SB, DIN, Wv_t, bv, nullptr, 0, vh, SB, Sn, DIN, DIN, Bn);
        // action sync -> q chain
        sync_update<<<(Bn * 512 + 255) / 256, 256, 0, stream>>>(act, li_a, ri_a, decay_a,
                                                                aaN, aaD, syncA_h);
        GEMM(syncA_h, 0, 512, Wqp_t, bqp, nullptr, 0, q1_h, 0, Bn, DIN, 512, 1);
        GEMM(q1_h, 0, DIN, Wq_t, bq, q2_f, 0, nullptr, 0, Bn, DIN, DIN, 1);
        // attention core (writes f16) + output projection
        attention<<<Bn * 8, 64, 0, stream>>>(q2_f, kh, vh, attno_h);
        GEMM(attno_h, 0, DIN, Wo_t, bo, attnp_f, 0, nullptr, 0, Bn, DIN, DIN, 1);
        // synapses: concat -> GEMM -> fused GLU+LN
        concat_pre<<<(Bn * 1536 + 255) / 256, 256, 0, stream>>>(attnp_f, act, pre_h);
        GEMM(pre_h, 0, 1536, Wsyn_t, bsyn, g_f, 0, nullptr, 0, Bn, 2048, 1536, 1);
        glu_ln<<<Bn, 256, 0, stream>>>(g_f, g_syn, b_syn, ph + (long)t * Bn * DMn);
        // neuron-level model -> new act
        nlm_kernel<<<DMn, 64, 0, stream>>>(init_hist, ph, t, W1, b1, W2, b2, act);
        // out sync -> prediction -> entropy
        sync_update<<<(Bn * 512 + 255) / 256, 256, 0, stream>>>(act, li_o, ri_o, decay_o,
                                                                aoN, aoD, syncO_h);
        GEMM(syncO_h, 0, 512, Wout_t, bout, pred_f, 0, nullptr, 0, Bn, DOUTn, 512, 1);
        finalize<<<Bn, 512, 0, stream>>>(pred_f, (float*)d_out, t);
    }
}
